// JambaMambaDecoderLayer_27736898797985
// MI455X (gfx1250) — compile-verified, matrix operand order checked by reading, not run on hardware
//
#include <hip/hip_runtime.h>
#include <hip/hip_bf16.h>
#include <math.h>

// Problem dims (fixed by the reference)
#define Bb     2
#define Ss     1024
#define TOK    2048      // B*S
#define Hdim   2048
#define DIdim  4096
#define Nst    16
#define Kc     4
#define Rr     128
#define FFd    5632

typedef __bf16 bf16;
typedef __attribute__((ext_vector_type(16))) __bf16 v16bf;
typedef __attribute__((ext_vector_type(8)))  __bf16 v8bf;
typedef __attribute__((ext_vector_type(8)))  float  v8f;

__device__ __forceinline__ float silu_f(float x) { return x / (1.0f + __expf(-x)); }

// CDNA5 async global->LDS copy (ASYNCcnt-tracked, bypasses VGPRs).
__device__ __forceinline__ void async_ld128(unsigned lds_addr, const void* gaddr) {
  asm volatile("global_load_async_to_lds_b128 %0, %1, off"
               :: "v"(lds_addr), "v"((unsigned long long)(uintptr_t)gaddr)
               : "memory");
}
__device__ __forceinline__ void wait_async_le6() {
  asm volatile("s_wait_asynccnt 0x6" ::: "memory");
}
__device__ __forceinline__ void wait_async_le0() {
  asm volatile("s_wait_asynccnt 0x0" ::: "memory");
}
__device__ __forceinline__ unsigned lds_off(const void* p) {
  // LDS aperture rule: flat addr[31:0] == LDS byte address.
  return (unsigned)(uintptr_t)p;
}

// ---------------------------------------------------------------------------
// fp32 -> bf16 conversion, 4 elements/thread (n must be a multiple of 4)
// ---------------------------------------------------------------------------
__global__ void cvt_f32_bf16_v4(const float4* __restrict__ s,
                                unsigned long long* __restrict__ d, long n4) {
  long i = (long)blockIdx.x * blockDim.x + threadIdx.x;
  if (i >= n4) return;
  float4 v = s[i];
  union { bf16 b[4]; unsigned long long u; } t;
  t.b[0] = (bf16)v.x; t.b[1] = (bf16)v.y; t.b[2] = (bf16)v.z; t.b[3] = (bf16)v.w;
  d[i] = t.u;
}

// ---------------------------------------------------------------------------
// Tiled WMMA GEMM:  C[M,N] = A[M,K](bf16,row-major) * W[N,K](bf16,row-major)^T
// Block tile 128x256, 8 waves (2x4), each wave 64x64 = 4x4 WMMA tiles, BK=32.
// Double-buffered LDS fed by GLOBAL_LOAD_ASYNC_TO_LDS_B128; last iteration
// peeled so the steady-state loop has no branches.
// Requires: M % 128 == 0, K % 32 == 0 (true for all call sites).
// Out-of-range W rows are clamped (their C columns are never stored).
// ---------------------------------------------------------------------------
#define BM 128
#define BN 256
#define BK 32
#define LDT 40   // padded LDS row stride (bf16 elements)

__global__ __launch_bounds__(256)
void gemm_bf16_nt(const bf16* __restrict__ A, const bf16* __restrict__ W,
                  float* __restrict__ C, int M, int N, int K)
{
  __shared__ bf16 As[2][BM * LDT];   // 2 x 10 KB
  __shared__ bf16 Ws[2][BN * LDT];   // 2 x 20 KB

  const int tid  = threadIdx.x;
  const int lane = tid & 31;
  const int wid  = tid >> 5;
  const int wm   = wid >> 2;      // 0..1
  const int wn   = wid & 3;       // 0..3
  const int r    = lane & 15;
  const int half = lane >> 4;
  const int blockM = blockIdx.y * BM;
  const int blockN = blockIdx.x * BN;

  // Staging: thread stages A rows (row0, row0+64) and W rows (row0+{0,64,128,192})
  // at column chunk c8 (8 bf16 = 16 B per async copy).
  const int row0 = tid >> 2;
  const int c8   = (tid & 3) << 3;

  const bf16* aG[2];
  aG[0] = A + (size_t)(blockM + row0)      * K + c8;
  aG[1] = A + (size_t)(blockM + row0 + 64) * K + c8;
  const bf16* wG[4];
#pragma unroll
  for (int q = 0; q < 4; ++q) {
    int wr = blockN + row0 + q * 64;
    if (wr >= N) wr = N - 1;                 // clamp: those C columns never stored
    wG[q] = W + (size_t)wr * K + c8;
  }

  unsigned aL[2][2], wL[2][4];
#pragma unroll
  for (int b = 0; b < 2; ++b) {
    aL[b][0] = lds_off(&As[b][row0 * LDT + c8]);
    aL[b][1] = lds_off(&As[b][(row0 + 64) * LDT + c8]);
#pragma unroll
    for (int q = 0; q < 4; ++q)
      wL[b][q] = lds_off(&Ws[b][(row0 + q * 64) * LDT + c8]);
  }

  const v8f vzero = {0.f, 0.f, 0.f, 0.f, 0.f, 0.f, 0.f, 0.f};
  v8f acc[4][4];
#pragma unroll
  for (int i = 0; i < 4; ++i)
#pragma unroll
    for (int j = 0; j < 4; ++j) acc[i][j] = vzero;

  union FR { v16bf v; v8bf h[2]; };

  auto issue = [&](int buf, int koff) {
#pragma unroll
    for (int q = 0; q < 2; ++q) async_ld128(aL[buf][q], aG[q] + koff);
#pragma unroll
    for (int q = 0; q < 4; ++q) async_ld128(wL[buf][q], wG[q] + koff);
  };

  auto compute = [&](int buf) {
    // Per-ISA 16-bit A-matrix layout: lane r holds row, halves split K
    // (half 0: K 0..7 & 16..23, half 1: K 8..15 & 24..31).
    FR af[4], wf[4];
    const int ka = half * 8;
#pragma unroll
    for (int tm = 0; tm < 4; ++tm) {
      int rowl = wm * 64 + tm * 16 + r;
      af[tm].h[0] = *(const v8bf*)&As[buf][rowl * LDT + ka];
      af[tm].h[1] = *(const v8bf*)&As[buf][rowl * LDT + ka + 16];
    }
#pragma unroll
    for (int tn = 0; tn < 4; ++tn) {
      int coll = wn * 64 + tn * 16 + r;
      wf[tn].h[0] = *(const v8bf*)&Ws[buf][coll * LDT + ka];
      wf[tn].h[1] = *(const v8bf*)&Ws[buf][coll * LDT + ka + 16];
    }
#pragma unroll
    for (int tm = 0; tm < 4; ++tm)
#pragma unroll
      for (int tn = 0; tn < 4; ++tn)
        acc[tm][tn] = __builtin_amdgcn_wmma_f32_16x16x32_bf16(
            false, af[tm].v, false, wf[tn].v, (short)0, acc[tm][tn], false, false);
  };

  const int nsteps = K / BK;

  // Prologue: stage k-step 0 into buffer 0.
  issue(0, 0);

  // Steady state: always issue next, wait for current, compute current.
  for (int i = 0; i < nsteps - 1; ++i) {
    const int cur = i & 1;
    __syncthreads();                 // buffer cur^1 free to refill
    issue(cur ^ 1, (i + 1) * BK);
    wait_async_le6();                // my 6 older copies (buffer cur) landed
    __syncthreads();                 // all waves' buffer-cur copies landed
    compute(cur);
  }
  // Epilogue: last K-step.
  __syncthreads();
  wait_async_le0();
  __syncthreads();
  compute((nsteps - 1) & 1);

  // C/D layout: lane r = column, VGPR v = row (+8 for upper half-lane group).
#pragma unroll
  for (int tm = 0; tm < 4; ++tm) {
    int rowBase = blockM + wm * 64 + tm * 16 + half * 8;
#pragma unroll
    for (int tn = 0; tn < 4; ++tn) {
      int col = blockN + wn * 64 + tn * 16 + r;
      if (col < N) {
#pragma unroll
        for (int v = 0; v < 8; ++v)
          C[(size_t)(rowBase + v) * N + col] = acc[tm][tn][v];
      }
    }
  }
}

// ---------------------------------------------------------------------------
// Generic RMSNorm over rows: dst = src * rsqrt(mean(src^2)+eps) * w
// ---------------------------------------------------------------------------
__global__ void rmsnorm_generic(const float* __restrict__ src, const float* __restrict__ w,
                                float* __restrict__ dst_f32, bf16* __restrict__ dst_b16,
                                int cols, int sstride, int dstride)
{
  __shared__ float red[128];
  const int row = blockIdx.x;
  const int tid = threadIdx.x;
  const float* sr = src + (size_t)row * sstride;
  float ss = 0.f;
  for (int c = tid; c < cols; c += 128) { float v = sr[c]; ss += v * v; }
  red[tid] = ss; __syncthreads();
  for (int off = 64; off > 0; off >>= 1) {
    if (tid < off) red[tid] += red[tid + off];
    __syncthreads();
  }
  float rs = rsqrtf(red[0] / (float)cols + 1e-6f);
  for (int c = tid; c < cols; c += 128) {
    float o = sr[c] * rs * w[c];
    if (dst_f32) dst_f32[(size_t)row * dstride + c] = o;
    if (dst_b16) dst_b16[(size_t)row * dstride + c] = (bf16)o;
  }
}

// ---------------------------------------------------------------------------
// Depthwise causal conv1d (K=4) + bias + SiLU on first DI columns of proj.
// ---------------------------------------------------------------------------
__global__ void conv_silu(const float* __restrict__ proj, const float* __restrict__ cw,
                          const float* __restrict__ cb, float* __restrict__ u32,
                          bf16* __restrict__ ub, long total)
{
  long i = (long)blockIdx.x * 256 + threadIdx.x;
  if (i >= total) return;
  int  d = (int)(i % DIdim);
  long t = i / DIdim;
  int  s = (int)(t % Ss);
  long b = t / Ss;
  float acc = cb[d];
#pragma unroll
  for (int k = 0; k < Kc; ++k) {
    int ss2 = s + k - (Kc - 1);
    if (ss2 >= 0)
      acc += cw[d * Kc + k] * proj[((b * Ss + ss2) * (size_t)(2 * DIdim)) + d];
  }
  float v = silu_f(acc);
  u32[i] = v;
  ub[i]  = (bf16)v;
}

// ---------------------------------------------------------------------------
// Selective scan, one thread per (b, d) channel, N=16 states in registers.
// Softplus(dt_raw + b_dt) fused in; B_t/C_t broadcast through double-buffered
// LDS (one barrier per step). Fuses D skip + SiLU(gate), writes bf16 y.
// ---------------------------------------------------------------------------
__global__ __launch_bounds__(256)
void scan_kernel(const float* __restrict__ u, const float* __restrict__ dtraw,
                 const float* __restrict__ bdt,
                 const float* __restrict__ Bn, const float* __restrict__ Cn,
                 const float* __restrict__ A_log, const float* __restrict__ Dsk,
                 const float* __restrict__ proj, bf16* __restrict__ yb)
{
  __shared__ float sB[2][Nst], sC[2][Nst];
  const int tid = threadIdx.x;
  const int nchunk = DIdim / 256;
  const int b = blockIdx.x / nchunk;
  const int d = (blockIdx.x % nchunk) * 256 + tid;

  float A[Nst], h[Nst];
#pragma unroll
  for (int n = 0; n < Nst; ++n) {
    A[n] = -__expf(A_log[(size_t)d * Nst + n]);
    h[n] = 0.f;
  }
  const float Dd  = Dsk[d];
  const float bdv = bdt[d];

  for (int s = 0; s < Ss; ++s) {
    size_t tok = (size_t)b * Ss + s;
    const int p = s & 1;
    if (tid < Nst)            sB[p][tid]       = Bn[tok * Nst + tid];
    else if (tid < 2 * Nst)   sC[p][tid - Nst] = Cn[tok * Nst + (tid - Nst)];
    __syncthreads();   // also fences re-write of this parity two steps later

    float dv  = dtraw[tok * DIdim + d] + bdv;
    float dtv = (dv > 20.f) ? dv : log1pf(__expf(dv));   // softplus
    float uv  = u[tok * DIdim + d];
    float y = 0.f;
#pragma unroll
    for (int n = 0; n < Nst; ++n) {
      float dA = __expf(dtv * A[n]);
      h[n] = dA * h[n] + dtv * sB[p][n] * uv;
      y += h[n] * sC[p][n];
    }
    y += uv * Dd;
    float g = proj[tok * (size_t)(2 * DIdim) + DIdim + d];
    y *= silu_f(g);
    yb[tok * DIdim + d] = (bf16)y;
  }
}

// ---------------------------------------------------------------------------
// hidden = res + mix (f32 kept for final residual); h_b = rmsnorm(hidden)*w
// cols fixed at Hdim=2048, 256 threads, 8 values/thread.
// ---------------------------------------------------------------------------
__global__ __launch_bounds__(256)
void add_rmsnorm(const float* __restrict__ res, const float* __restrict__ mix,
                 const float* __restrict__ w, float* __restrict__ hidden,
                 bf16* __restrict__ hb)
{
  __shared__ float red[256];
  const int row = blockIdx.x, tid = threadIdx.x;
  float lv[8];
  float ss = 0.f;
#pragma unroll
  for (int i = 0; i < 8; ++i) {
    int c = tid + i * 256;
    float v = res[(size_t)row * Hdim + c] + mix[(size_t)row * Hdim + c];
    hidden[(size_t)row * Hdim + c] = v;
    lv[i] = v; ss += v * v;
  }
  red[tid] = ss; __syncthreads();
  for (int off = 128; off > 0; off >>= 1) {
    if (tid < off) red[tid] += red[tid + off];
    __syncthreads();
  }
  float rs = rsqrtf(red[0] / (float)Hdim + 1e-6f);
#pragma unroll
  for (int i = 0; i < 8; ++i) {
    int c = tid + i * 256;
    hb[(size_t)row * Hdim + c] = (bf16)(lv[i] * rs * w[c]);
  }
}

// ---------------------------------------------------------------------------
// ff_act = silu(gate) * up from the fused [gate|up] GEMM output -> bf16
// gu layout: [TOK, 2*FFd]; gate at cols [0,FFd), up at cols [FFd, 2*FFd).
// ---------------------------------------------------------------------------
__global__ void silu_mul_fused(const float* __restrict__ gu, bf16* __restrict__ o, long n)
{
  long i = (long)blockIdx.x * 256 + threadIdx.x;
  if (i >= n) return;
  long row = i / FFd;
  int  c   = (int)(i % FFd);
  float g = gu[row * (size_t)(2 * FFd) + c];
  float u = gu[row * (size_t)(2 * FFd) + FFd + c];
  o[i] = (bf16)(silu_f(g) * u);
}

// ---------------------------------------------------------------------------
// out = hidden + ff
// ---------------------------------------------------------------------------
__global__ void add_out(const float* __restrict__ a, const float* __restrict__ b,
                        float* __restrict__ o, long n)
{
  long i = (long)blockIdx.x * 256 + threadIdx.x;
  if (i < n) o[i] = a[i] + b[i];
}

// ---------------------------------------------------------------------------
// Host launcher
// ---------------------------------------------------------------------------
extern "C" void kernel_launch(void* const* d_in, const int* in_sizes, int n_in,
                              void* d_out, int out_size, void* d_ws, size_t ws_size,
                              hipStream_t stream)
{
  (void)in_sizes; (void)n_in; (void)out_size; (void)ws_size;

  const float* hs    = (const float*)d_in[0];
  const float* ilnw  = (const float*)d_in[1];
  const float* w_in  = (const float*)d_in[2];
  const float* cw    = (const float*)d_in[3];
  const float* cb    = (const float*)d_in[4];
  const float* w_x   = (const float*)d_in[5];
  const float* dtln  = (const float*)d_in[6];
  const float* bln   = (const float*)d_in[7];
  const float* cln   = (const float*)d_in[8];
  const float* w_dt  = (const float*)d_in[9];
  const float* bdt   = (const float*)d_in[10];
  const float* alog  = (const float*)d_in[11];
  const float* dsk   = (const float*)d_in[12];
  const float* w_out = (const float*)d_in[13];
  const float* pfln  = (const float*)d_in[14];
  const float* w_g   = (const float*)d_in[15];
  const float* w_u   = (const float*)d_in[16];
  const float* w_d   = (const float*)d_in[17];

  size_t off = 0;
  auto alloc = [&](size_t elems, size_t esz) -> void* {
    void* p = (char*)d_ws + off;
    off += (elems * esz + 255) & ~(size_t)255;
    return p;
  };

  bf16*  w_in_b  = (bf16*)alloc((size_t)2 * DIdim * Hdim, 2);
  bf16*  w_x_b   = (bf16*)alloc((size_t)(Rr + 2 * Nst) * DIdim, 2);
  bf16*  w_dt_b  = (bf16*)alloc((size_t)DIdim * Rr, 2);
  bf16*  w_out_b = (bf16*)alloc((size_t)Hdim * DIdim, 2);
  bf16*  w_gu_b  = (bf16*)alloc((size_t)2 * FFd * Hdim, 2);   // [gate ; up]
  bf16*  w_d_b   = (bf16*)alloc((size_t)Hdim * FFd, 2);
  bf16*  xn_b    = (bf16*)alloc((size_t)TOK * Hdim, 2);
  float* proj    = (float*)alloc((size_t)TOK * 2 * DIdim, 4);
  float* u32     = (float*)alloc((size_t)TOK * DIdim, 4);
  bf16*  u_b     = (bf16*)alloc((size_t)TOK * DIdim, 2);
  float* ssmp    = (float*)alloc((size_t)TOK * (Rr + 2 * Nst), 4);
  bf16*  tr_b    = (bf16*)alloc((size_t)TOK * Rr, 2);
  float* Bn      = (float*)alloc((size_t)TOK * Nst, 4);
  float* Cn      = (float*)alloc((size_t)TOK * Nst, 4);
  float* dtbuf   = (float*)alloc((size_t)TOK * DIdim, 4);
  bf16*  y_b     = (bf16*)alloc((size_t)TOK * DIdim, 2);
  float* mix     = (float*)alloc((size_t)TOK * Hdim, 4);
  float* hidden  = (float*)alloc((size_t)TOK * Hdim, 4);
  bf16*  h_b     = (bf16*)alloc((size_t)TOK * Hdim, 2);
  float* gubuf   = (float*)alloc((size_t)TOK * 2 * FFd, 4);
  bf16*  ff_b    = (bf16*)alloc((size_t)TOK * FFd, 2);
  float* ffout   = (float*)alloc((size_t)TOK * Hdim, 4);

  auto cdiv = [](long a, long b) -> long { return (a + b - 1) / b; };

  // --- Weight conversions (fp32 -> bf16), 4 elems/thread ---
  {
    long n;
    n = (long)2 * DIdim * Hdim;
    cvt_f32_bf16_v4<<<cdiv(n / 4, 256), 256, 0, stream>>>((const float4*)w_in,  (unsigned long long*)w_in_b,  n / 4);
    n = (long)(Rr + 2 * Nst) * DIdim;
    cvt_f32_bf16_v4<<<cdiv(n / 4, 256), 256, 0, stream>>>((const float4*)w_x,   (unsigned long long*)w_x_b,   n / 4);
    n = (long)DIdim * Rr;
    cvt_f32_bf16_v4<<<cdiv(n / 4, 256), 256, 0, stream>>>((const float4*)w_dt,  (unsigned long long*)w_dt_b,  n / 4);
    n = (long)Hdim * DIdim;
    cvt_f32_bf16_v4<<<cdiv(n / 4, 256), 256, 0, stream>>>((const float4*)w_out, (unsigned long long*)w_out_b, n / 4);
    n = (long)FFd * Hdim;
    cvt_f32_bf16_v4<<<cdiv(n / 4, 256), 256, 0, stream>>>((const float4*)w_g,   (unsigned long long*)w_gu_b,  n / 4);
    cvt_f32_bf16_v4<<<cdiv(n / 4, 256), 256, 0, stream>>>((const float4*)w_u,   (unsigned long long*)(w_gu_b + (size_t)FFd * Hdim), n / 4);
    n = (long)Hdim * FFd;
    cvt_f32_bf16_v4<<<cdiv(n / 4, 256), 256, 0, stream>>>((const float4*)w_d,   (unsigned long long*)w_d_b,   n / 4);
  }

  // --- Input RMSNorm -> bf16 activations ---
  rmsnorm_generic<<<TOK, 128, 0, stream>>>(hs, ilnw, nullptr, xn_b, Hdim, Hdim, Hdim);

  // --- in-proj: proj[2048, 8192] = xn @ w_in^T ---
  {
    dim3 g((unsigned)cdiv(2 * DIdim, BN), (unsigned)cdiv(TOK, BM));
    gemm_bf16_nt<<<g, 256, 0, stream>>>(xn_b, w_in_b, proj, TOK, 2 * DIdim, Hdim);
  }

  // --- causal conv + SiLU on u half ---
  {
    long n = (long)TOK * DIdim;
    conv_silu<<<cdiv(n, 256), 256, 0, stream>>>(proj, cw, cb, u32, u_b, n);
  }

  // --- ssm-proj: ssmp[2048, 160] = u @ w_x^T ---
  {
    dim3 g((unsigned)cdiv(Rr + 2 * Nst, BN), (unsigned)cdiv(TOK, BM));
    gemm_bf16_nt<<<g, 256, 0, stream>>>(u_b, w_x_b, ssmp, TOK, Rr + 2 * Nst, DIdim);
  }

  // --- split + RMSNorms of t_r / B / C ---
  rmsnorm_generic<<<TOK, 128, 0, stream>>>(ssmp,            dtln, nullptr, tr_b, Rr,  Rr + 2 * Nst, Rr);
  rmsnorm_generic<<<TOK, 128, 0, stream>>>(ssmp + Rr,       bln,  Bn, nullptr, Nst, Rr + 2 * Nst, Nst);
  rmsnorm_generic<<<TOK, 128, 0, stream>>>(ssmp + Rr + Nst, cln,  Cn, nullptr, Nst, Rr + 2 * Nst, Nst);

  // --- dt-proj: dtbuf[2048, 4096] = t_r @ w_dt^T (softplus fused into scan) ---
  {
    dim3 g((unsigned)cdiv(DIdim, BN), (unsigned)cdiv(TOK, BM));
    gemm_bf16_nt<<<g, 256, 0, stream>>>(tr_b, w_dt_b, dtbuf, TOK, DIdim, Rr);
  }

  // --- selective scan (fused softplus + D skip + SiLU(gate)) -> y bf16 ---
  scan_kernel<<<Bb * (DIdim / 256), 256, 0, stream>>>(u32, dtbuf, bdt, Bn, Cn, alog, dsk, proj, y_b);

  // --- out-proj: mix[2048, 2048] = y @ w_out^T ---
  {
    dim3 g((unsigned)cdiv(Hdim, BN), (unsigned)cdiv(TOK, BM));
    gemm_bf16_nt<<<g, 256, 0, stream>>>(y_b, w_out_b, mix, TOK, Hdim, DIdim);
  }

  // --- residual add + pre-FF RMSNorm ---
  add_rmsnorm<<<TOK, 256, 0, stream>>>(hs, mix, pfln, hidden, h_b);

  // --- FF fused gate|up GEMM: gubuf[2048, 11264] = h @ [w_g; w_u]^T ---
  {
    dim3 g((unsigned)cdiv(2 * FFd, BN), (unsigned)cdiv(TOK, BM));
    gemm_bf16_nt<<<g, 256, 0, stream>>>(h_b, w_gu_b, gubuf, TOK, 2 * FFd, Hdim);
    long n = (long)TOK * FFd;
    silu_mul_fused<<<cdiv(n, 256), 256, 0, stream>>>(gubuf, ff_b, n);
  }

  // --- FF down GEMM + final residual ---
  {
    dim3 g((unsigned)cdiv(Hdim, BN), (unsigned)cdiv(TOK, BM));
    gemm_bf16_nt<<<g, 256, 0, stream>>>(ff_b, w_d_b, ffout, TOK, Hdim, FFd);
    long n = (long)TOK * Hdim;
    add_out<<<cdiv(n, 256), 256, 0, stream>>>(hidden, ffout, (float*)d_out, n);
  }
}
